// LinearQwen3OmniMoeThinkerTextSparseMoeBlock_2181843386484
// MI455X (gfx1250) — matrix operand order repaired and checked by reading, not compiled, not run on hardware
//
#include <hip/hip_runtime.h>
#include <hip/hip_bf16.h>
#include <math.h>

#define T    2048
#define Hdim 1024
#define NE   32
#define NI   768

typedef __attribute__((ext_vector_type(16))) __bf16 v16bf;
typedef __attribute__((ext_vector_type(8)))  float  v8f;

// ---------------------------------------------------------------------------
// Kernel 1: router. One wave (32 lanes) per token.
//   logits = x @ router_w.T  -> fp32 softmax -> top-4 -> renorm
// ---------------------------------------------------------------------------
__global__ __launch_bounds__(256) void router_topk_kernel(
    const float* __restrict__ x, const float* __restrict__ rw,
    int* __restrict__ tidx, float* __restrict__ tw) {
  int wave = threadIdx.x >> 5, lane = threadIdx.x & 31;
  int t = blockIdx.x * 8 + wave;
  const float* xr = x + (size_t)t * Hdim;

  float logit = 0.f;
  for (int e = 0; e < NE; ++e) {
    const float* wr = rw + (size_t)e * Hdim;
    float acc = 0.f;
    for (int h = lane; h < Hdim; h += 32) acc = fmaf(xr[h], wr[h], acc);
    #pragma unroll
    for (int off = 16; off; off >>= 1) acc += __shfl_xor(acc, off, 32);
    if (lane == e) logit = acc;
  }
  float m = logit;
  #pragma unroll
  for (int off = 16; off; off >>= 1) m = fmaxf(m, __shfl_xor(m, off, 32));
  float p = __expf(logit - m);
  float s = p;
  #pragma unroll
  for (int off = 16; off; off >>= 1) s += __shfl_xor(s, off, 32);
  p /= s;
  float vals[4]; int eidx[4];
  float cur = p;
  #pragma unroll
  for (int k = 0; k < 4; ++k) {
    float v = cur; int ei = lane;
    #pragma unroll
    for (int off = 16; off; off >>= 1) {
      float ov = __shfl_xor(v, off, 32);
      int   oe = __shfl_xor(ei, off, 32);
      if (ov > v || (ov == v && oe < ei)) { v = ov; ei = oe; }
    }
    vals[k] = v; eidx[k] = ei;
    if (lane == ei) cur = -1.f;
  }
  float sum4 = vals[0] + vals[1] + vals[2] + vals[3];
  if (lane < 4) {
    tidx[t * 4 + lane] = eidx[lane];
    tw[t * 4 + lane]   = vals[lane] / sum4;
  }
}

// ---------------------------------------------------------------------------
// Kernel 2: deterministic per-expert token compaction (ballot prefix, no
// atomics -> stable ordering across replays).
// ---------------------------------------------------------------------------
__global__ __launch_bounds__(32) void build_lists_kernel(
    const int* __restrict__ tidx, int* __restrict__ counts,
    int* __restrict__ list) {
  int e = blockIdx.x, lane = threadIdx.x;
  int base = 0;
  for (int t0 = 0; t0 < T; t0 += 32) {
    int t = t0 + lane;
    int found = -1;
    #pragma unroll
    for (int k = 0; k < 4; ++k)
      if (tidx[t * 4 + k] == e) found = k;
    unsigned long long mask = __ballot(found >= 0);
    if (found >= 0) {
      int pos = base + (int)__popcll(mask & ((1ull << lane) - 1ull));
      list[(size_t)e * T + pos] = t * 4 + found;
    }
    base += (int)__popcll(mask);
  }
  if (lane == 0) counts[e] = base;
}

// ---------------------------------------------------------------------------
// Kernel 3: fused expert MLP. Block = (expert e, 32-token tile), 16 waves.
// Two M-tiles per block so every B-fragment (weight) load feeds 2 WMMAs.
//  Phase A: [32x768] gate/up via bf16 WMMA over K=H; SiLU(g)*u -> LDS bf16
//  Phase B: [32x768] @ down^T -> [32x1024], scale by routing weight,
//           global_atomic_add_f32 scatter into out.
// A-frag: lane m holds row_m K = half*16 .. +15 (contiguous, matches B-frag)
// B-frag: lane n holds w[n0+n, kb+half*16 .. +15] (row-major, no transpose)
// ---------------------------------------------------------------------------
__global__ __launch_bounds__(512) void expert_mlp_kernel(
    const float* __restrict__ x,  const float* __restrict__ gw,
    const float* __restrict__ uw, const float* __restrict__ dw,
    const float* __restrict__ tw, const int* __restrict__ list,
    const int* __restrict__ counts, float* __restrict__ out) {
  int e = blockIdx.y;
  int tile = blockIdx.x;
  int cnt = counts[e];
  if (tile * 32 >= cnt) return;   // uniform exit; EXEC all-ones below

  __shared__ int    rows_t[32];
  __shared__ float  rows_w[32];
  __shared__ __bf16 Hlds[32 * NI];   // 48 KB hidden tile

  int tid = threadIdx.x, wave = tid >> 5, lane = tid & 31;
  int m16 = lane & 15, half = lane >> 4;

  if (tid < 32) {
    int i = tile * 32 + tid;
    if (i < cnt) {
      int slot = list[(size_t)e * T + i];
      rows_t[tid] = slot >> 2;
      rows_w[tid] = tw[slot];
    } else {
      rows_t[tid] = 0;
      rows_w[tid] = 0.f;   // pad rows contribute 0
    }
  }
  __syncthreads();

  int row0 = rows_t[m16];
  int row1 = rows_t[16 + m16];
  v8f zero = {};
  v8f accg[2][3], accu[2][3];
  #pragma unroll
  for (int t = 0; t < 3; ++t) {
    accg[0][t] = zero; accg[1][t] = zero;
    accu[0][t] = zero; accu[1][t] = zero;
  }

  const float* xb0 = x + (size_t)row0 * Hdim + half * 16;
  const float* xb1 = x + (size_t)row1 * Hdim + half * 16;

  // ---- Phase A: gate & up GEMMs, K = H = 1024 ----
  for (int kb = 0; kb < Hdim; kb += 32) {
    v16bf a0, a1;
    #pragma unroll
    for (int j = 0; j < 8; ++j) {
      a0[j]     = (__bf16)xb0[kb + j];
      a0[8 + j] = (__bf16)xb0[kb + 8 + j];
      a1[j]     = (__bf16)xb1[kb + j];
      a1[8 + j] = (__bf16)xb1[kb + 8 + j];
    }
    #pragma unroll
    for (int t = 0; t < 3; ++t) {
      int n0 = wave * 48 + t * 16 + m16;
      const float* gp = gw + ((size_t)e * NI + n0) * Hdim + kb + half * 16;
      const float* up = uw + ((size_t)e * NI + n0) * Hdim + kb + half * 16;
      __builtin_prefetch(gp + 32, 0, 0);
      __builtin_prefetch(up + 32, 0, 0);
      v16bf bg, bu;
      #pragma unroll
      for (int j = 0; j < 16; ++j) {
        bg[j] = (__bf16)gp[j];
        bu[j] = (__bf16)up[j];
      }
      accg[0][t] = __builtin_amdgcn_wmma_f32_16x16x32_bf16(
          false, a0, false, bg, (short)0, accg[0][t], false, false);
      accg[1][t] = __builtin_amdgcn_wmma_f32_16x16x32_bf16(
          false, a1, false, bg, (short)0, accg[1][t], false, false);
      accu[0][t] = __builtin_amdgcn_wmma_f32_16x16x32_bf16(
          false, a0, false, bu, (short)0, accu[0][t], false, false);
      accu[1][t] = __builtin_amdgcn_wmma_f32_16x16x32_bf16(
          false, a1, false, bu, (short)0, accu[1][t], false, false);
    }
  }

  // SiLU(g) * u -> Hlds.  C layout: N = m16(+base), M = vgpr + half*8.
  #pragma unroll
  for (int mt = 0; mt < 2; ++mt) {
    #pragma unroll
    for (int t = 0; t < 3; ++t) {
      int n = wave * 48 + t * 16 + m16;
      #pragma unroll
      for (int r = 0; r < 8; ++r) {
        int m = mt * 16 + r + half * 8;
        float g = accg[mt][t][r];
        float hval = (g / (1.f + __expf(-g))) * accu[mt][t][r];
        Hlds[m * NI + n] = (__bf16)hval;
      }
    }
  }
  __syncthreads();

  // ---- Phase B: down GEMM, K = I = 768, N = H = 1024 ----
  v8f accd[2][4];
  #pragma unroll
  for (int t = 0; t < 4; ++t) { accd[0][t] = zero; accd[1][t] = zero; }

  for (int kb = 0; kb < NI; kb += 32) {
    const __bf16* hb0 = &Hlds[m16 * NI + kb + half * 16];
    const __bf16* hb1 = &Hlds[(16 + m16) * NI + kb + half * 16];
    v16bf a0, a1;
    #pragma unroll
    for (int j = 0; j < 8; ++j) {
      a0[j]     = hb0[j];
      a0[8 + j] = hb0[8 + j];
      a1[j]     = hb1[j];
      a1[8 + j] = hb1[8 + j];
    }
    #pragma unroll
    for (int t = 0; t < 4; ++t) {
      int n0 = wave * 64 + t * 16 + m16;
      const float* dp = dw + ((size_t)e * Hdim + n0) * NI + kb + half * 16;
      __builtin_prefetch(dp + 32, 0, 0);
      v16bf bd;
      #pragma unroll
      for (int j = 0; j < 16; ++j) bd[j] = (__bf16)dp[j];
      accd[0][t] = __builtin_amdgcn_wmma_f32_16x16x32_bf16(
          false, a0, false, bd, (short)0, accd[0][t], false, false);
      accd[1][t] = __builtin_amdgcn_wmma_f32_16x16x32_bf16(
          false, a1, false, bd, (short)0, accd[1][t], false, false);
    }
  }

  // scale by routing weight, scatter-accumulate (4 experts per token)
  #pragma unroll
  for (int mt = 0; mt < 2; ++mt) {
    #pragma unroll
    for (int t = 0; t < 4; ++t) {
      int h = wave * 64 + t * 16 + m16;
      #pragma unroll
      for (int r = 0; r < 8; ++r) {
        int m = mt * 16 + r + half * 8;
        float v = accd[mt][t][r] * rows_w[m];
        atomicAdd(&out[(size_t)rows_t[m] * Hdim + h], v);
      }
    }
  }
}

// ---------------------------------------------------------------------------
extern "C" void kernel_launch(void* const* d_in, const int* in_sizes, int n_in,
                              void* d_out, int out_size, void* d_ws,
                              size_t ws_size, hipStream_t stream) {
  const float* x  = (const float*)d_in[0];  // [1, 2048, 1024]
  const float* rw = (const float*)d_in[1];  // [32, 1024]
  const float* gw = (const float*)d_in[2];  // [32, 768, 1024]
  const float* uw = (const float*)d_in[3];  // [32, 768, 1024]
  const float* dw = (const float*)d_in[4];  // [32, 1024, 768]
  float* out = (float*)d_out;               // [1, 2048, 1024]

  char* ws = (char*)d_ws;
  int*   counts   = (int*)ws;                                   // 32 ints
  int*   topk_idx = (int*)(ws + 256);                           // T*4 ints
  float* topk_w   = (float*)(ws + 256 + (size_t)T * 4 * 4);     // T*4 floats
  int*   list     = (int*)(ws + 256 + (size_t)T * 4 * 8);       // E*T ints

  hipMemsetAsync(out, 0, (size_t)T * Hdim * sizeof(float), stream);

  router_topk_kernel<<<T / 8, 256, 0, stream>>>(x, rw, topk_idx, topk_w);
  build_lists_kernel<<<NE, 32, 0, stream>>>(topk_idx, counts, list);
  expert_mlp_kernel<<<dim3(T / 32, NE), 512, 0, stream>>>(
      x, gw, uw, dw, topk_w, list, counts, out);
}
